// DeepARS4_51616916963812
// MI455X (gfx1250) — matrix operand order repaired and checked by reading, not compiled, not run on hardware
//
#include <hip/hip_runtime.h>

// ---------------- problem constants ----------------
#define Bb 16
#define Ll 2048
#define DIN 8
#define Hh 256
#define Nn 64
#define NL 4
#define BL (Bb * Ll)          // 32768 rows

typedef unsigned short ushort_t;
typedef __attribute__((ext_vector_type(16))) __bf16 v16bf;
typedef __attribute__((ext_vector_type(8)))  float  v8f;

union FragAB { uint4 q[2]; v16bf v; };

static __device__ __forceinline__ ushort_t f2bf(float f) {
    union { float f; unsigned u; } v; v.f = f;
    unsigned r = (v.u + 0x7FFFu + ((v.u >> 16) & 1u)) >> 16;   // RNE
    return (ushort_t)r;
}
static __device__ __forceinline__ float softplus_f(float x) {
    return fmaxf(x, 0.f) + log1pf(expf(-fabsf(x)));
}
// CDNA5 async global->LDS copy (ASYNCcnt), 16B per lane, no VGPR round-trip
static __device__ __forceinline__ void async_b128(unsigned lds_off, const void* gaddr) {
    asm volatile("global_load_async_to_lds_b128 %0, %1, off"
                 :: "v"(lds_off), "v"(gaddr) : "memory");
}
static __device__ __forceinline__ unsigned lds_off_of(const void* p) {
    return (unsigned)(size_t)p;   // low 32 bits of generic LDS address = LDS offset
}

// ---------------- encoder: x(BL,8) @ enc_w.T(8,H) + b ----------------
__global__ void enc_kernel(const float* __restrict__ x, const float* __restrict__ ew,
                           const float* __restrict__ eb, float* __restrict__ xcur) {
    int row = blockIdx.x;         // 0..BL-1
    int h   = threadIdx.x;        // 0..255
    const float* xr = x + (size_t)row * DIN;
    float acc = eb[h];
#pragma unroll
    for (int d = 0; d < DIN; ++d) acc = fmaf(xr[d], ew[h * DIN + d], acc);
    xcur[(size_t)row * Hh + h] = acc;
}

// ---------------- per-layer SSM discretization ----------------
__global__ void ssm_param_kernel(const float* __restrict__ log_dt, const float* __restrict__ logA_re,
                                 const float* __restrict__ A_im,  const float* __restrict__ C_re,
                                 const float* __restrict__ C_im,
                                 float* __restrict__ wr, float* __restrict__ wi,
                                 float* __restrict__ cr, float* __restrict__ ci) {
    int idx = blockIdx.x * blockDim.x + threadIdx.x;   // 0..H*N-1
    int h   = idx >> 6;
    float dt  = expf(log_dt[h]);
    float are = -expf(logA_re[idx]);
    float aim = A_im[idx];
    float e   = expf(are * dt);
    float w_r = e * cosf(aim * dt);
    float w_i = e * sinf(aim * dt);
    float inv = 1.f / (are * are + aim * aim);
    float nr  = w_r - 1.f, ni = w_i;
    float qr  = (nr * are + ni * aim) * inv;           // (w-1)/A = (w-1)*conj(A)/|A|^2
    float qi  = (ni * are - nr * aim) * inv;
    float CR  = C_re[idx], CI = C_im[idx];
    wr[idx] = w_r; wi[idx] = w_i;
    cr[idx] = 2.f * (CR * qr - CI * qi);
    ci[idx] = 2.f * (CR * qi + CI * qr);
}

// ---------------- SSM recurrence + D-skip + GELU -> bf16 ----------------
// one wave = two (b,h) sequences; 16 lanes * 4 complex states = N=64
__global__ void scan_gelu_kernel(const float* __restrict__ xcur,
                                 const float* __restrict__ wr, const float* __restrict__ wi,
                                 const float* __restrict__ cr, const float* __restrict__ ci,
                                 const float* __restrict__ Dp, ushort_t* __restrict__ ybf) {
    int lane = threadIdx.x & 31;
    int wavi = threadIdx.x >> 5;
    int grp  = lane >> 4;          // which of 2 sequences
    int gl   = lane & 15;          // lane inside group
    int seq  = (blockIdx.x * 8 + wavi) * 2 + grp;       // 0..B*H-1
    int b = seq >> 8, h = seq & 255;

    int base = h * Nn + gl * 4;
    float wrv[4], wiv[4], crv[4], civ[4], sr[4], si[4];
#pragma unroll
    for (int j = 0; j < 4; ++j) {
        wrv[j] = wr[base + j]; wiv[j] = wi[base + j];
        crv[j] = cr[base + j]; civ[j] = ci[base + j];
        sr[j] = 0.f; si[j] = 0.f;
    }
    float dsk = Dp[h];
    const float*  up = xcur + (size_t)b * Ll * Hh + h;
    ushort_t*     yp = ybf  + (size_t)b * Ll * Hh + h;

    for (int t0 = 0; t0 < Ll; t0 += 16) {
        float uvec = up[(size_t)(t0 + gl) * Hh];   // 16 future inputs, one per lane
        float yout = 0.f;
#pragma unroll
        for (int s = 0; s < 16; ++s) {
            float u = __shfl(uvec, (grp << 4) + s, 32);
            float p = 0.f;
#pragma unroll
            for (int j = 0; j < 4; ++j) {
                float tr = fmaf(wrv[j], sr[j], fmaf(-wiv[j], si[j], u));
                float ti = fmaf(wrv[j], si[j],  wiv[j] * sr[j]);
                sr[j] = tr; si[j] = ti;
                p = fmaf(crv[j], tr, fmaf(-civ[j], ti, p));
            }
            p += __shfl_xor(p, 1, 32);
            p += __shfl_xor(p, 2, 32);
            p += __shfl_xor(p, 4, 32);
            p += __shfl_xor(p, 8, 32);
            float y = fmaf(dsk, u, p);
            float gy = 0.5f * y * (1.f + erff(y * 0.70710678118f));   // exact GELU
            if (gl == s) yout = gy;
        }
        yp[(size_t)(t0 + gl) * Hh] = f2bf(yout);
    }
}

// ---------------- f32 -> bf16 weight convert ----------------
__global__ void f2bf_kernel(const float* __restrict__ src, ushort_t* __restrict__ dst, int n) {
    int i = blockIdx.x * blockDim.x + threadIdx.x;
    if (i < n) dst[i] = f2bf(src[i]);
}

// ---------------- WMMA bf16 GEMM + bias + fused GLU ----------------
// zz = A(BL,256) @ W.T(256,512) + b ; g = zz[:, :256] * sigmoid(zz[:, 256:])
// grid: (BL/64, 256/64); block 256 = 8 waves; wave tile 16M x 32 GLU-cols
// Double-buffered LDS tiles filled by global_load_async_to_lds_b128 (ASYNCcnt),
// overlapping the async copy of chunk k+1 with WMMA on chunk k.
__global__ __launch_bounds__(256) void gemm_glu_kernel(const ushort_t* __restrict__ Abf,
                                                       const ushort_t* __restrict__ Wbf,
                                                       const float* __restrict__ bias,
                                                       float* __restrict__ g) {
    __shared__ ushort_t As[2][64][72];    // padded rows to de-conflict banks
    __shared__ ushort_t Bs[2][128][72];

    int tid  = threadIdx.x;
    int lane = tid & 31;
    int wid  = tid >> 5;
    int wm   = wid >> 1, wn = wid & 1;
    int rowBase = blockIdx.x * 64;
    int gnBase  = blockIdx.y * 64;

    // per-thread staging coordinates (64x64 A tile: 16 bf16/thread; 128x64 B tile: 32 bf16/thread)
    int ra = tid >> 2, ca = (tid & 3) * 16;
    int rb = tid >> 1, cb = (tid & 1) * 32;
    int jg = (rb < 64) ? (gnBase + rb) : (256 + gnBase + (rb - 64));
    const ushort_t* gA = Abf + (size_t)(rowBase + ra) * Hh + ca;
    const ushort_t* gB = Wbf + (size_t)jg * Hh + cb;

    auto issue_chunk = [&](int kb, int bs) {
        const ushort_t* pa = gA + kb * 64;
        async_b128(lds_off_of(&As[bs][ra][ca]),     pa);
        async_b128(lds_off_of(&As[bs][ra][ca + 8]), pa + 8);
        const ushort_t* pb = gB + kb * 64;
        async_b128(lds_off_of(&Bs[bs][rb][cb]),      pb);
        async_b128(lds_off_of(&Bs[bs][rb][cb + 8]),  pb + 8);
        async_b128(lds_off_of(&Bs[bs][rb][cb + 16]), pb + 16);
        async_b128(lds_off_of(&Bs[bs][rb][cb + 24]), pb + 24);
    };

    v8f aL0 = {}, aL1 = {}, aR0 = {}, aR1 = {};

    issue_chunk(0, 0);
#pragma unroll
    for (int kb = 0; kb < 4; ++kb) {
        int bs = kb & 1;
        if (kb < 3) {
            issue_chunk(kb + 1, bs ^ 1);                     // prefetch next chunk
            asm volatile("s_wait_asynccnt 0x6" ::: "memory"); // chunk kb landed (in-order), next 6 in flight
        } else {
            asm volatile("s_wait_asynccnt 0x0" ::: "memory");
        }
        __syncthreads();

#pragma unroll
        for (int ks = 0; ks < 2; ++ks) {
            int k0 = ks * 32;
            int kh = lane >> 4;
            int m  = wm * 16 + (lane & 15);
            FragAB a, bL0, bL1, bR0, bR1;
            // A frag: 16 contiguous K per lane-half (K in {kh*16..kh*16+15})
            a.q[0] = *(const uint4*)&As[bs][m][k0 + kh * 16];
            a.q[1] = *(const uint4*)&As[bs][m][k0 + kh * 16 + 8];
            // B frag: lane=column, 16 contiguous K per lane-half
            int n  = lane & 15;
            int jl = wn * 32 + n;
            const uint4* p;
            p = (const uint4*)&Bs[bs][jl][k0 + kh * 16];           bL0.q[0] = p[0]; bL0.q[1] = p[1];
            p = (const uint4*)&Bs[bs][jl + 16][k0 + kh * 16];      bL1.q[0] = p[0]; bL1.q[1] = p[1];
            p = (const uint4*)&Bs[bs][64 + jl][k0 + kh * 16];      bR0.q[0] = p[0]; bR0.q[1] = p[1];
            p = (const uint4*)&Bs[bs][64 + jl + 16][k0 + kh * 16]; bR1.q[0] = p[0]; bR1.q[1] = p[1];

            aL0 = __builtin_amdgcn_wmma_f32_16x16x32_bf16(false, a.v, false, bL0.v, (short)0, aL0, false, false);
            aL1 = __builtin_amdgcn_wmma_f32_16x16x32_bf16(false, a.v, false, bL1.v, (short)0, aL1, false, false);
            aR0 = __builtin_amdgcn_wmma_f32_16x16x32_bf16(false, a.v, false, bR0.v, (short)0, aR0, false, false);
            aR1 = __builtin_amdgcn_wmma_f32_16x16x32_bf16(false, a.v, false, bR1.v, (short)0, aR1, false, false);
        }
        __syncthreads();   // all waves done reading buffer bs before it is refilled
    }

    // epilogue: bias + GLU, scatter per C/D layout (vgpr r -> M=r or 8+r)
    int n     = lane & 15;
    int rsel  = (lane < 16) ? 0 : 8;
    int colL0 = gnBase + wn * 32 + n;
    int colL1 = colL0 + 16;
    float bL0 = bias[colL0],       bL1 = bias[colL1];
    float bR0 = bias[colL0 + 256], bR1 = bias[colL1 + 256];
#pragma unroll
    for (int r = 0; r < 8; ++r) {
        int row = rowBase + wm * 16 + rsel + r;
        float zL0 = aL0[r] + bL0, zR0 = aR0[r] + bR0;
        float zL1 = aL1[r] + bL1, zR1 = aR1[r] + bR1;
        g[(size_t)row * Hh + colL0] = zL0 / (1.f + expf(-zR0));
        g[(size_t)row * Hh + colL1] = zL1 / (1.f + expf(-zR1));
    }
}

// ---------------- residual + layernorm (in-place on xcur) ----------------
__global__ void res_ln_kernel(const float* __restrict__ g, float* __restrict__ xcur,
                              const float* __restrict__ gam, const float* __restrict__ bet) {
    int lane = threadIdx.x & 31;
    int row  = blockIdx.x * 8 + (threadIdx.x >> 5);
    const float* gp = g    + (size_t)row * Hh;
    float*       xp = xcur + (size_t)row * Hh;
    float v[8], s = 0.f, s2 = 0.f;
#pragma unroll
    for (int j = 0; j < 8; ++j) {
        int c = lane + 32 * j;
        float t = gp[c] + xp[c];
        v[j] = t; s += t; s2 = fmaf(t, t, s2);
    }
#pragma unroll
    for (int m = 1; m < 32; m <<= 1) { s += __shfl_xor(s, m, 32); s2 += __shfl_xor(s2, m, 32); }
    float mean = s * (1.f / Hh);
    float var  = s2 * (1.f / Hh) - mean * mean;
    float rstd = rsqrtf(var + 1e-5f);
#pragma unroll
    for (int j = 0; j < 8; ++j) {
        int c = lane + 32 * j;
        xp[c] = fmaf((v[j] - mean) * rstd, gam[c], bet[c]);
    }
}

// ---------------- heads: softplus(x @ w.T + b) ----------------
__global__ void head_kernel(const float* __restrict__ xcur,
                            const float* __restrict__ muw, const float* __restrict__ mub,
                            const float* __restrict__ alw, const float* __restrict__ alb,
                            float* __restrict__ out) {
    int lane = threadIdx.x & 31;
    int row  = blockIdx.x * 8 + (threadIdx.x >> 5);
    const float* xp = xcur + (size_t)row * Hh;
    float sm = 0.f, sa = 0.f;
#pragma unroll
    for (int j = 0; j < 8; ++j) {
        int c = lane + 32 * j;
        float t = xp[c];
        sm = fmaf(t, muw[c], sm);
        sa = fmaf(t, alw[c], sa);
    }
#pragma unroll
    for (int m = 1; m < 32; m <<= 1) { sm += __shfl_xor(sm, m, 32); sa += __shfl_xor(sa, m, 32); }
    if (lane == 0) {
        out[row]      = softplus_f(sm + mub[0]);
        out[BL + row] = softplus_f(sa + alb[0]);
    }
}

// ---------------- launcher ----------------
extern "C" void kernel_launch(void* const* d_in, const int* in_sizes, int n_in,
                              void* d_out, int out_size, void* d_ws, size_t ws_size,
                              hipStream_t stream) {
    const float* x       = (const float*)d_in[0];
    const float* enc_w   = (const float*)d_in[1];
    const float* enc_b   = (const float*)d_in[2];
    const float* log_dt  = (const float*)d_in[3];
    const float* logA_re = (const float*)d_in[4];
    const float* A_im    = (const float*)d_in[5];
    const float* C_re    = (const float*)d_in[6];
    const float* C_im    = (const float*)d_in[7];
    const float* Dp      = (const float*)d_in[8];
    const float* out_w   = (const float*)d_in[9];
    const float* out_b   = (const float*)d_in[10];
    const float* ln_g    = (const float*)d_in[11];
    const float* ln_b    = (const float*)d_in[12];
    const float* mu_w    = (const float*)d_in[13];
    const float* mu_b    = (const float*)d_in[14];
    const float* al_w    = (const float*)d_in[15];
    const float* al_b    = (const float*)d_in[16];

    char* ws = (char*)d_ws;
    size_t off = 0;
    float*    xcur = (float*)(ws + off);    off += (size_t)BL * Hh * 4;    // 32 MB
    float*    gbuf = (float*)(ws + off);    off += (size_t)BL * Hh * 4;    // 32 MB
    ushort_t* ybf  = (ushort_t*)(ws + off); off += (size_t)BL * Hh * 2;    // 16 MB
    ushort_t* wbf  = (ushort_t*)(ws + off); off += (size_t)2 * Hh * Hh * 2;
    float*    wr   = (float*)(ws + off);    off += (size_t)Hh * Nn * 4;
    float*    wi   = (float*)(ws + off);    off += (size_t)Hh * Nn * 4;
    float*    cr   = (float*)(ws + off);    off += (size_t)Hh * Nn * 4;
    float*    ci   = (float*)(ws + off);    off += (size_t)Hh * Nn * 4;

    enc_kernel<<<BL, 256, 0, stream>>>(x, enc_w, enc_b, xcur);

    for (int i = 0; i < NL; ++i) {
        const size_t hn = (size_t)Hh * Nn;
        ssm_param_kernel<<<(Hh * Nn) / 256, 256, 0, stream>>>(
            log_dt + (size_t)i * Hh, logA_re + i * hn, A_im + i * hn,
            C_re + i * hn, C_im + i * hn, wr, wi, cr, ci);

        scan_gelu_kernel<<<(Bb * Hh) / 16, 256, 0, stream>>>(xcur, wr, wi, cr, ci,
                                                             Dp + (size_t)i * Hh, ybf);

        f2bf_kernel<<<(2 * Hh * Hh) / 256, 256, 0, stream>>>(out_w + (size_t)i * 2 * Hh * Hh,
                                                             wbf, 2 * Hh * Hh);

        dim3 gg(BL / 64, Hh / 64);   // 512 x 4
        gemm_glu_kernel<<<gg, 256, 0, stream>>>(ybf, wbf, out_b + (size_t)i * 2 * Hh, gbuf);

        res_ln_kernel<<<BL / 8, 256, 0, stream>>>(gbuf, xcur,
                                                  ln_g + (size_t)i * Hh, ln_b + (size_t)i * Hh);
    }

    head_kernel<<<BL / 8, 256, 0, stream>>>(xcur, mu_w, mu_b, al_w, al_b, (float*)d_out);
}
